// HGTRegressor_91268055040643
// MI455X (gfx1250) — matrix operand order repaired and checked by reading, not compiled, not run on hardware
//
#include <hip/hip_runtime.h>
#include <hip/hip_bf16.h>
#include <math.h>

// ---------------- problem constants ----------------
#define Hh   4
#define Dd   32
#define HID  128
#define NA   100000
#define NW   20000
#define NO   50000
#define NTOT 170000
#define Ecnt 150000
#define KTOT 270000   // sum of source-node counts over the 4 edge types

// ---------------- WMMA types ----------------
typedef __attribute__((ext_vector_type(16))) __bf16 v16bf;
typedef __attribute__((ext_vector_type(2)))  __bf16 v2bf;
typedef __attribute__((ext_vector_type(8)))  float  v8f;

__device__ inline unsigned pack_bf16(float x, float y) {
  v2bf v;
  v[0] = (__bf16)x;
  v[1] = (__bf16)y;
  return __builtin_bit_cast(unsigned, v);   // -> single v_cvt_pk_bf16_f32
}

// =====================================================================
// Tiled GEMM:  C[M,N] = act(A[M,K] @ B[K,N] + bias)
// fp32 in memory -> bf16 staged in LDS **in WMMA fragment order** so each
// lane reads its whole 16-element operand as one contiguous 32B LDS load.
// Block: 256 threads = 8 waves, tile 64(M) x BN_(N), K stepped by 32.
//   BN_=64: wave -> 1 A-frag reused for 2 WMMAs (mt=wave&3, 2 n-tiles)
//   BN_=32: wave -> 1 WMMA (mt=wave>>1, nt=wave&1)
// Requires N % BN_ == 0 and K % 32 == 0 (true for all uses here).
// OOB M rows: A row index is CLAMPED (not zeroed) — a garbage A row only
// affects the same (never-stored) D row, so no guarded-load path is needed.
// Fragment-order LDS layout (ISA 7.12.2, 16-bit A 16x32 / B 32x16):
//   A elem (m,k): lane = (m&15) + 16*((k>>3)&1), elem = (k&7) + 8*(k>>4)
//   B elem (k,n): lane = (n&15) + 16*((k>>3)&1), elem = (k&7) + 8*(k>>4)
// act: 0 = none, 1 = relu
// =====================================================================
#define TBM 64

template<int BN_>
__global__ __launch_bounds__(256) void wmma_gemm_kernel(
    const float* __restrict__ A, int lda,
    const float* __restrict__ B, int ldb,
    const float* __restrict__ bias,
    float* __restrict__ C, int ldc,
    int M, int N, int K, int act)
{
  constexpr int NTILES = BN_ / 16;   // 16-wide n-tiles in block
  constexpr int TPW    = BN_ / 32;   // n-tiles per wave (1 or 2)

  __shared__ unsigned As_u[4][32][8];        // 4 m-tiles, frag order, 4KB
  __shared__ unsigned Bs_u[NTILES][32][8];   // 2/4KB

  const int tid  = threadIdx.x;
  const int lane = tid & 31;
  const int wave = tid >> 5;
  const int lo   = lane & 15;
  const int hi   = lane >> 4;
  const int mt   = (BN_ == 32) ? (wave >> 1) : (wave & 3);
  const int ntb  = (BN_ == 32) ? (wave & 1) : ((wave >> 2) * TPW);

  const int m0 = blockIdx.x * TBM;
  const int n0 = blockIdx.y * BN_;
  const bool mfull = (m0 + TBM <= M);

  v8f acc[TPW];
  #pragma unroll
  for (int j = 0; j < TPW; ++j) acc[j] = (v8f){0.f,0.f,0.f,0.f,0.f,0.f,0.f,0.f};

  for (int k0 = 0; k0 < K; k0 += 32) {
    // ---- stage A (64x32): unconditional float4 load, row clamped ----
    // idx: r = row (0..63), p2 = k/4 (0..7)
    #pragma unroll
    for (int it = 0; it < 2; ++it) {
      int idx = tid + it * 256;               // 0..511
      int r = idx >> 3, p2 = idx & 7;
      int k = 4 * p2;
      int alane = (r & 15) + 16 * ((p2 >> 1) & 1);
      int e     = 2 * (p2 & 1) + 4 * (p2 >> 2);
      int grow  = m0 + r;
      if (grow >= M) grow = M - 1;            // clamp: garbage rows unsaved
      float4 a4 = *(const float4*)(A + (size_t)grow * lda + (k0 + k));
      uint2 pk = make_uint2(pack_bf16(a4.x, a4.y), pack_bf16(a4.z, a4.w));
      *(uint2*)&As_u[r >> 4][alane][e] = pk;
    }
    // ---- stage B (32 x BN_): two float4 rows -> 4 packed dwords ----
    // idx: kp = k-pair (0..15), n4 = n/4
    for (int idx = tid; idx < BN_ * 4; idx += 256) {
      int kp = idx / (BN_ / 4);
      int n  = 4 * (idx % (BN_ / 4));
      const float* bp = B + (size_t)(k0 + 2 * kp) * ldb + (n0 + n);
      float4 r0 = *(const float4*)bp;
      float4 r1 = *(const float4*)(bp + ldb);
      int blane0 = 16 * ((kp >> 2) & 1);
      int e      = (kp & 3) + 4 * (kp >> 3);
      Bs_u[(n + 0) >> 4][blane0 + ((n + 0) & 15)][e] = pack_bf16(r0.x, r1.x);
      Bs_u[(n + 1) >> 4][blane0 + ((n + 1) & 15)][e] = pack_bf16(r0.y, r1.y);
      Bs_u[(n + 2) >> 4][blane0 + ((n + 2) & 15)][e] = pack_bf16(r0.z, r1.z);
      Bs_u[(n + 3) >> 4][blane0 + ((n + 3) & 15)][e] = pack_bf16(r0.w, r1.w);
    }
    __syncthreads();

    // ---- one contiguous 32B LDS read per operand, then WMMA ----
    v16bf af = *reinterpret_cast<v16bf*>(&As_u[mt][lane][0]);
    #pragma unroll
    for (int j = 0; j < TPW; ++j) {
      v16bf bfb = *reinterpret_cast<v16bf*>(&Bs_u[ntb + j][lane][0]);
      acc[j] = __builtin_amdgcn_wmma_f32_16x16x32_bf16(
          false, af, false, bfb, (short)0, acc[j], false, false);
    }
    __syncthreads();
  }

  // ---- D layout: vgpr r -> M = r (lanes 0-15) / r+8 (lanes 16-31) ----
  #pragma unroll
  for (int j = 0; j < TPW; ++j) {
    int col = n0 + (ntb + j) * 16 + lo;
    float bv = bias ? bias[col] : 0.f;
    #pragma unroll
    for (int r = 0; r < 8; ++r) {
      int row = m0 + mt * 16 + r + hi * 8;
      float v = acc[j][r] + bv;
      if (act == 1) v = fmaxf(v, 0.f);
      if (mfull || row < M) C[(size_t)row * ldc + col] = v;
    }
  }
}

// =====================================================================
// helpers
// =====================================================================
__device__ inline void atomic_max_f32(float* addr, float v) {
  if (v >= 0.f) atomicMax((int*)addr, __float_as_int(v));
  else          atomicMin((unsigned int*)addr, __float_as_uint(v));
}
__device__ inline void atomic_add_f32(float* p, float v) {
  unsafeAtomicAdd(p, v);   // hardware global_atomic_add_f32
}

__global__ void fill_kernel(float* __restrict__ p, float v, int n) {
  int i = blockIdx.x * blockDim.x + threadIdx.x;
  if (i < n) p[i] = v;
}

__global__ void gelu_kernel(float* __restrict__ p, int n) {
  int i = blockIdx.x * blockDim.x + threadIdx.x;
  if (i < n) {
    float x = p[i];
    p[i] = 0.5f * x * (1.0f + erff(x * 0.70710678118654752f));
  }
}

// =====================================================================
// alpha pass: one wave per edge.  lane l -> head h=l/8, dims 4*(l%8)..+3
// =====================================================================
__global__ __launch_bounds__(256) void alpha_kernel(
    const int* __restrict__ src, const int* __restrict__ dst,
    const float* __restrict__ ke,      // [Nsrc][128]
    const float* __restrict__ q,       // rows stride 384, at q columns
    const float* __restrict__ prel,    // [H]
    float* __restrict__ alpha_out,     // [E][H]
    float* __restrict__ amax,          // [NTOT][H]
    int dst_off, int nE)
{
  int wid  = (blockIdx.x * blockDim.x + threadIdx.x) >> 5;
  int lane = threadIdx.x & 31;
  if (wid >= nE) return;
  int s = src[wid], d = dst[wid];
  const float4* qr = (const float4*)(q  + (size_t)d * 384);
  const float4* kr = (const float4*)(ke + (size_t)s * 128);
  float4 qv = qr[lane];
  float4 kv = kr[lane];
  float part = qv.x*kv.x + qv.y*kv.y + qv.z*kv.z + qv.w*kv.w;
  part += __shfl_xor(part, 1, 32);
  part += __shfl_xor(part, 2, 32);
  part += __shfl_xor(part, 4, 32);
  int h = lane >> 3;
  if ((lane & 7) == 0) {
    float a = part * prel[h] * 0.17677669529663687f;  // 1/sqrt(32)
    alpha_out[(size_t)wid * Hh + h] = a;
    atomic_max_f32(&amax[(size_t)(dst_off + d) * Hh + h], a);
  }
}

// ex = exp(alpha - amax[dstg]);  den[dstg] += ex   (alpha buffer reused)
__global__ void exden_kernel(
    const int* __restrict__ dst, float* __restrict__ alpha,
    const float* __restrict__ amax, float* __restrict__ den,
    int dst_off, int nE)
{
  int i = blockIdx.x * blockDim.x + threadIdx.x;
  if (i >= nE * Hh) return;
  int e = i >> 2, h = i & 3;
  size_t gi = (size_t)(dst_off + dst[e]) * Hh + h;
  float ex = expf(alpha[i] - amax[gi]);
  alpha[i] = ex;
  atomic_add_f32(&den[gi], ex);
}

// agg[dstg, h*32+d] += (ex/(den+1e-16)) * ve[src, h*32+d];  wave per edge
__global__ __launch_bounds__(256) void agg_kernel(
    const int* __restrict__ src, const int* __restrict__ dst,
    const float* __restrict__ ve, const float* __restrict__ ex,
    const float* __restrict__ den, float* __restrict__ agg,
    int dst_off, int nE)
{
  int wid  = (blockIdx.x * blockDim.x + threadIdx.x) >> 5;
  int lane = threadIdx.x & 31;
  if (wid >= nE) return;
  int s = src[wid], d = dst[wid];
  int h = lane >> 3;
  size_t gi = (size_t)(dst_off + d) * Hh + h;
  float w = ex[(size_t)wid * Hh + h] / (den[gi] + 1e-16f);
  const float4* vr = (const float4*)(ve + (size_t)s * 128);
  float4 v = vr[lane];
  float* b = agg + (size_t)(dst_off + d) * 128 + lane * 4;
  atomic_add_f32(b + 0, w * v.x);
  atomic_add_f32(b + 1, w * v.y);
  atomic_add_f32(b + 2, w * v.z);
  atomic_add_f32(b + 3, w * v.w);
}

// =====================================================================
// post: y = g*o + (1-g)*x ; x = relu(LN(y)*ln_g + ln_b)  (wave per row)
// =====================================================================
__global__ __launch_bounds__(256) void post_kernel(
    const float* __restrict__ o, float* __restrict__ x,
    const float* __restrict__ skip_p,
    const float* __restrict__ lng, const float* __restrict__ lnb,
    int nrows)
{
  int wid  = (blockIdx.x * blockDim.x + threadIdx.x) >> 5;
  int lane = threadIdx.x & 31;
  if (wid >= nrows) return;
  float g = 1.f / (1.f + expf(-skip_p[0]));
  const float4* orow = (const float4*)(o + (size_t)wid * 128);
  float4*       xrow = (float4*)(x + (size_t)wid * 128);
  float4 ov = orow[lane], xv = xrow[lane];
  float4 y;
  y.x = g * ov.x + (1.f - g) * xv.x;
  y.y = g * ov.y + (1.f - g) * xv.y;
  y.z = g * ov.z + (1.f - g) * xv.z;
  y.w = g * ov.w + (1.f - g) * xv.w;
  float s  = y.x + y.y + y.z + y.w;
  float s2 = y.x*y.x + y.y*y.y + y.z*y.z + y.w*y.w;
  #pragma unroll
  for (int m = 1; m < 32; m <<= 1) {
    s  += __shfl_xor(s,  m, 32);
    s2 += __shfl_xor(s2, m, 32);
  }
  float mu  = s * (1.f / 128.f);
  float var = s2 * (1.f / 128.f) - mu * mu;
  float inv = rsqrtf(var + 1e-5f);
  float4 gg = ((const float4*)lng)[lane];
  float4 bb = ((const float4*)lnb)[lane];
  float4 r;
  r.x = fmaxf((y.x - mu) * inv * gg.x + bb.x, 0.f);
  r.y = fmaxf((y.y - mu) * inv * gg.y + bb.y, 0.f);
  r.z = fmaxf((y.z - mu) * inv * gg.z + bb.z, 0.f);
  r.w = fmaxf((y.w - mu) * inv * gg.w + bb.w, 0.f);
  xrow[lane] = r;
}

// out[i] = dot(x[i,:], w_head) + b_head + base   (wave per row)
__global__ __launch_bounds__(256) void head_kernel(
    const float* __restrict__ x, const float* __restrict__ w,
    const float* __restrict__ bh, const float* __restrict__ base,
    float* __restrict__ out, int nrows)
{
  int wid  = (blockIdx.x * blockDim.x + threadIdx.x) >> 5;
  int lane = threadIdx.x & 31;
  if (wid >= nrows) return;
  float4 xv = ((const float4*)(x + (size_t)wid * 128))[lane];
  float4 wv = ((const float4*)w)[lane];
  float p = xv.x*wv.x + xv.y*wv.y + xv.z*wv.z + xv.w*wv.w;
  #pragma unroll
  for (int m = 1; m < 32; m <<= 1) p += __shfl_xor(p, m, 32);
  if (lane == 0) out[wid] = p + bh[0] + base[0];
}

// =====================================================================
// host side
// =====================================================================
static inline void gemm(const float* A, int lda, const float* B, int ldb,
                        const float* bias, float* C, int ldc,
                        int M, int N, int K, int act, hipStream_t s)
{
  if (N % 64 == 0) {
    dim3 g((M + TBM - 1) / TBM, N / 64), b(256);
    wmma_gemm_kernel<64><<<g, b, 0, s>>>(A, lda, B, ldb, bias, C, ldc, M, N, K, act);
  } else {
    dim3 g((M + TBM - 1) / TBM, N / 32), b(256);
    wmma_gemm_kernel<32><<<g, b, 0, s>>>(A, lda, B, ldb, bias, C, ldc, M, N, K, act);
  }
}
static inline void fill(float* p, float v, int n, hipStream_t s) {
  fill_kernel<<<(n + 255) / 256, 256, 0, s>>>(p, v, n);
}

extern "C" void kernel_launch(void* const* d_in, const int* in_sizes, int n_in,
                              void* d_out, int out_size, void* d_ws, size_t ws_size,
                              hipStream_t stream)
{
  (void)in_sizes; (void)n_in; (void)out_size; (void)ws_size;
  const float* x_t[3]  = {(const float*)d_in[0], (const float*)d_in[1], (const float*)d_in[2]};
  const float* W_in    = (const float*)d_in[3];
  const float* b_in    = (const float*)d_in[4];
  const float* W_kqv   = (const float*)d_in[5];
  const float* b_kqv   = (const float*)d_in[6];
  const float* a_rel   = (const float*)d_in[7];
  const float* m_rel   = (const float*)d_in[8];
  const float* p_rel   = (const float*)d_in[9];
  const float* skip_p  = (const float*)d_in[10];
  const float* W_o     = (const float*)d_in[11];
  const float* b_o     = (const float*)d_in[12];
  const float* ln_g    = (const float*)d_in[13];
  const float* ln_b    = (const float*)d_in[14];
  const float* w_head  = (const float*)d_in[15];
  const float* b_head  = (const float*)d_in[16];
  const float* base    = (const float*)d_in[17];
  const int* srcs[4] = {(const int*)d_in[18], (const int*)d_in[20],
                        (const int*)d_in[22], (const int*)d_in[24]};
  const int* dsts[4] = {(const int*)d_in[19], (const int*)d_in[21],
                        (const int*)d_in[23], (const int*)d_in[25]};
  float* out = (float*)d_out;

  const int Nt[3]   = {NA, NW, NO};
  const int OFFS[4] = {0, NA, NA + NW, NA + NW + NO};
  const int st[4]   = {1, 2, 0, 0};   // source node type per edge type
  const int dt[4]   = {0, 0, 1, 2};   // dest node type per edge type
  const int KOFF[4] = {0, NW, NW + NO, NW + NO + NA};
  const int Ksrc[4] = {NW, NO, NA, NA};

  // ---------- workspace layout (floats, ~814 MB total) ----------
  float* ws = (float*)d_ws;
  size_t off = 0;
  float* buf_x    = ws + off; off += (size_t)NTOT * HID;       // node features
  float* buf_kqv  = ws + off; off += (size_t)NTOT * 3 * HID;   // k|q|v per node
  float* buf_ke   = ws + off; off += (size_t)KTOT * HID;       // rel-transformed K
  float* buf_ve   = ws + off; off += (size_t)KTOT * HID;       // rel-transformed V
  float* buf_al   = ws + off; off += (size_t)4 * Ecnt * Hh;    // alpha -> ex
  float* buf_amax = ws + off; off += (size_t)NTOT * Hh;
  float* buf_den  = ws + off; off += (size_t)NTOT * Hh;
  float* buf_agg  = ws + off; off += (size_t)NTOT * HID;
  float* buf_o    = ws + off; off += (size_t)NTOT * HID;

  const int edge_blocks = (Ecnt + 7) / 8;   // 8 waves / block, wave per edge

  // ---------- input projection: x = relu(x_in @ W_in[t] + b_in[t]) ----------
  for (int t = 0; t < 3; ++t)
    gemm(x_t[t], 64, W_in + (size_t)t * 64 * HID, HID, b_in + t * HID,
         buf_x + (size_t)OFFS[t] * HID, HID, Nt[t], HID, 64, /*relu*/1, stream);

  // ---------- 2 HGT layers ----------
  for (int l = 0; l < 2; ++l) {
    // kqv GEMMs
    for (int t = 0; t < 3; ++t)
      gemm(buf_x + (size_t)OFFS[t] * HID, HID,
           W_kqv + (size_t)(l * 3 + t) * HID * 3 * HID, 3 * HID,
           b_kqv + (size_t)(l * 3 + t) * 3 * HID,
           buf_kqv + (size_t)OFFS[t] * 3 * HID, 3 * HID,
           Nt[t], 3 * HID, HID, 0, stream);

    // per-head rel transforms (K = 32 -> exactly one WMMA step, BN_=32 path)
    for (int e = 0; e < 4; ++e)
      for (int h = 0; h < Hh; ++h) {
        const float* Ak = buf_kqv + (size_t)OFFS[st[e]] * 384 + h * Dd;
        const float* Av = buf_kqv + (size_t)OFFS[st[e]] * 384 + 2 * HID + h * Dd;
        const float* Ba = a_rel + (size_t)((l * 4 + e) * Hh + h) * Dd * Dd;
        const float* Bm = m_rel + (size_t)((l * 4 + e) * Hh + h) * Dd * Dd;
        gemm(Ak, 384, Ba, Dd, nullptr,
             buf_ke + (size_t)KOFF[e] * HID + h * Dd, HID,
             Ksrc[e], Dd, Dd, 0, stream);
        gemm(Av, 384, Bm, Dd, nullptr,
             buf_ve + (size_t)KOFF[e] * HID + h * Dd, HID,
             Ksrc[e], Dd, Dd, 0, stream);
      }

    // segment softmax buffers
    fill(buf_amax, -INFINITY, NTOT * Hh, stream);
    fill(buf_den, 0.f, NTOT * Hh, stream);
    fill(buf_agg, 0.f, NTOT * HID, stream);

    // alpha + segment max
    for (int e = 0; e < 4; ++e)
      alpha_kernel<<<edge_blocks, 256, 0, stream>>>(
          srcs[e], dsts[e], buf_ke + (size_t)KOFF[e] * HID,
          buf_kqv + (size_t)OFFS[dt[e]] * 384 + HID,   // q columns
          p_rel + (size_t)(l * 4 + e) * Hh,
          buf_al + (size_t)e * Ecnt * Hh, buf_amax, OFFS[dt[e]], Ecnt);

    // exp + segment sum
    for (int e = 0; e < 4; ++e)
      exden_kernel<<<(Ecnt * Hh + 255) / 256, 256, 0, stream>>>(
          dsts[e], buf_al + (size_t)e * Ecnt * Hh, buf_amax, buf_den,
          OFFS[dt[e]], Ecnt);

    // weighted V aggregation
    for (int e = 0; e < 4; ++e)
      agg_kernel<<<edge_blocks, 256, 0, stream>>>(
          srcs[e], dsts[e], buf_ve + (size_t)KOFF[e] * HID,
          buf_al + (size_t)e * Ecnt * Hh, buf_den, buf_agg,
          OFFS[dt[e]], Ecnt);

    // gelu (exact) in place
    gelu_kernel<<<(NTOT * HID + 255) / 256, 256, 0, stream>>>(buf_agg, NTOT * HID);

    // output projection
    for (int t = 0; t < 3; ++t)
      gemm(buf_agg + (size_t)OFFS[t] * HID, HID,
           W_o + (size_t)(l * 3 + t) * HID * HID, HID,
           b_o + (size_t)(l * 3 + t) * HID,
           buf_o + (size_t)OFFS[t] * HID, HID, Nt[t], HID, HID, 0, stream);

    // skip + layernorm + relu (updates buf_x in place)
    for (int t = 0; t < 3; ++t)
      post_kernel<<<(Nt[t] + 7) / 8, 256, 0, stream>>>(
          buf_o + (size_t)OFFS[t] * HID, buf_x + (size_t)OFFS[t] * HID,
          skip_p + (l * 3 + t),
          ln_g + (size_t)(l * 3 + t) * HID, ln_b + (size_t)(l * 3 + t) * HID,
          Nt[t]);
  }

  // ---------- regression head over agents ----------
  head_kernel<<<(NA + 7) / 8, 256, 0, stream>>>(
      buf_x, w_head, b_head, base, out, NA);
}
